// fea_59536836657810
// MI455X (gfx1250) — matrix operand order repaired and checked by reading, not compiled, not run on hardware
//
#include <hip/hip_runtime.h>
#include <math.h>

typedef __attribute__((ext_vector_type(16))) __bf16 v16bf;
typedef __attribute__((ext_vector_type(8)))  __bf16 v8bf;
typedef __attribute__((ext_vector_type(8)))  float  v8f;

#define BN_EPS 1e-5f
#define NEG_SLOPE 0.2f

// ---------------- workspace layout (bytes) ----------------
static constexpr size_t SUP_OFF    = 0;                 // support, <= 100000*128*4 = 51.2 MB
static constexpr size_t ACT_OFF    = (size_t)52 << 20;  // activated agg, <= 51.2 MB
static constexpr size_t RSTART_OFF = (size_t)104 << 20; // (N+1) ints
static constexpr size_t PART_OFF   = (size_t)105 << 20; // BN partials, <= 256 KB
static constexpr size_t STATS_OFF  = (size_t)106 << 20; // mean + rstd, <= 1 KB

// ---------------- CSR row offsets from sorted COO ----------------
__global__ void row_bounds(const int* __restrict__ row, int* __restrict__ rstart,
                           int N, int E) {
    int r = blockIdx.x * blockDim.x + threadIdx.x;
    if (r > N) return;
    if (r == N) { rstart[N] = E; return; }
    int lo = 0, hi = E;
    while (lo < hi) { int mid = (lo + hi) >> 1; if (row[mid] < r) lo = mid + 1; else hi = mid; }
    rstart[r] = lo;
}

// ---------------- GEMM: S[N,DOUT] = H[N,DIN] @ W, via v_wmma_f32_16x16x32_bf16 ----
// One wave computes a 16-row strip x all DOUT columns (COLT 16x16 tiles).
// W (f32) is staged ONCE per block into LDS in exact B-fragment layout (bf16):
//   fragment f = kc*COLT+ct occupies 1 KB; chunk c (elems 8c..8c+7) of lane l
//   lives at bf16 index  f*512 + c*256 + l*8  -> ds_load_b128 sweeps banks minimally.
template <int DIN, int DOUT>
__global__ __launch_bounds__(256) void gemm_wmma(const float* __restrict__ H,
                                                 const float* __restrict__ W,
                                                 float* __restrict__ S, int nRowTiles) {
    constexpr int COLT = DOUT / 16;
    constexpr int KC   = DIN / 32;
    constexpr int TOT  = DIN * DOUT;       // multiple of 256 for all layers

    __shared__ __bf16 sh[TOT];

    // ---- cooperative stage: W[k, col] (f32) -> fragment-layout bf16 in LDS ----
    for (int L = threadIdx.x; L < TOT; L += 256) {
        int f   = L >> 9;          // L / 512
        int r   = L & 511;
        int c   = r >> 8;          // chunk: elems 8c..8c+7
        int r2  = r & 255;
        int l   = r2 >> 3;         // lane
        int ec  = r2 & 7;
        int e   = c * 8 + ec;                       // 0..15 within K-chunk window
        int kc  = f / COLT, ct = f % COLT;
        int k   = kc * 32 + (l >> 4) * 8 + (e & 7) + (e >> 3) * 16;  // B layout: lane half h holds K = 8h..8h+7, 16+8h..16+8h+7
        int colI = ct * 16 + (l & 15);
        sh[L] = (__bf16)W[(size_t)k * DOUT + colI];
    }
    __syncthreads();

    const int wave = threadIdx.x >> 5;
    const int lane = threadIdx.x & 31;
    const int rt   = blockIdx.x * 8 + wave;
    if (rt >= nRowTiles) return;          // wave-uniform; after barrier, EXEC all-ones

    const int m    = lane & 15;
    const int half = lane >> 4;
    const float*  hrow = H + (size_t)(rt * 16 + m) * DIN;
    const __bf16* lp   = sh + lane * 8;   // this lane's slot within each chunk

    v8f acc[COLT] = {};

    for (int kc = 0; kc < KC; ++kc) {
        // A fragment (16x32 bf16): lane m holds K = kb..kb+7 and kb+16..kb+23
        const int kb = kc * 32 + half * 8;
        v16bf a;
#pragma unroll
        for (int j = 0; j < 8; ++j) {
            a[j]     = (__bf16)hrow[kb + j];
            a[8 + j] = (__bf16)hrow[kb + 16 + j];
        }

#pragma unroll
        for (int ct = 0; ct < COLT; ++ct) {
            const __bf16* fp = lp + (kc * COLT + ct) * 512;
            v8bf lo = *(const v8bf*)fp;           // ds_load_b128
            v8bf hi = *(const v8bf*)(fp + 256);   // ds_load_b128
            v16bf b = __builtin_shufflevector(lo, hi,
                        0, 1, 2, 3, 4, 5, 6, 7, 8, 9, 10, 11, 12, 13, 14, 15);
            acc[ct] = __builtin_amdgcn_wmma_f32_16x16x32_bf16(
                false, a, false, b, (short)0, acc[ct], false, false);
        }
    }

    // D layout: vgpr v, lane l -> M = v + 8*(l>>4), N = l&15
    float* srow = S + (size_t)rt * 16 * DOUT;
#pragma unroll
    for (int ct = 0; ct < COLT; ++ct)
#pragma unroll
        for (int v = 0; v < 8; ++v)
            srow[(size_t)(v + half * 8) * DOUT + ct * 16 + m] = acc[ct][v];
}

// ---------------- SpMM + fused elu(leaky_relu(x)) ----------------
template <int DOUT>
__global__ void spmm_act(const float* __restrict__ S, const int* __restrict__ col,
                         const float* __restrict__ vals, const int* __restrict__ rstart,
                         float* __restrict__ A) {
    const int r = blockIdx.x;
    const int d = threadIdx.x;
    const int s = rstart[r], e = rstart[r + 1];
    float sum = 0.f;
    for (int i = s; i < e; ++i) {
        sum += S[(size_t)col[i] * DOUT + d] * vals[i];
    }
    // elu(leaky_relu(x)): x>0 -> x ; x<=0 -> exp(0.2x)-1
    A[(size_t)r * DOUT + d] = (sum > 0.f) ? sum : (expf(NEG_SLOPE * sum) - 1.f);
}

// ---------------- BatchNorm: deterministic two-stage column stats ----------------
template <int DOUT>
__global__ __launch_bounds__(256) void bn_partial(const float* __restrict__ A,
                                                  float* __restrict__ P,
                                                  int N, int nChunks) {
    __shared__ float sh[256], sh2[256];
    constexpr int RG = 256 / DOUT;
    const int chunk = blockIdx.x;
    const int rowsPer = (N + nChunks - 1) / nChunks;
    const int r0 = chunk * rowsPer;
    const int r1 = min(N, r0 + rowsPer);
    const int c = threadIdx.x % DOUT;
    const int g = threadIdx.x / DOUT;
    float s = 0.f, s2 = 0.f;
    for (int r = r0 + g; r < r1; r += RG) {
        float v = A[(size_t)r * DOUT + c];
        s += v; s2 += v * v;
    }
    sh[threadIdx.x] = s; sh2[threadIdx.x] = s2;
    __syncthreads();
    if (g == 0) {
        for (int j = 1; j < RG; ++j) { s += sh[j * DOUT + c]; s2 += sh2[j * DOUT + c]; }
        P[(size_t)chunk * DOUT + c]             = s;
        P[(size_t)(nChunks + chunk) * DOUT + c] = s2;
    }
}

template <int DOUT>
__global__ void bn_finalize(const float* __restrict__ P, float* __restrict__ stats,
                            int N, int nChunks) {
    int c = threadIdx.x;
    if (c >= DOUT) return;
    float s = 0.f, s2 = 0.f;
    for (int k = 0; k < nChunks; ++k) {
        s  += P[(size_t)k * DOUT + c];
        s2 += P[(size_t)(nChunks + k) * DOUT + c];
    }
    float mean = s / (float)N;
    float var  = s2 / (float)N - mean * mean;   // biased var, matches jnp.var
    stats[c]        = mean;
    stats[DOUT + c] = rsqrtf(var + BN_EPS);
}

template <int DOUT>
__global__ void bn_apply(const float* __restrict__ A, const float* __restrict__ stats,
                         float* __restrict__ Out, int total) {
    int idx = blockIdx.x * blockDim.x + threadIdx.x;
    if (idx >= total) return;
    int c = idx % DOUT;
    Out[idx] = (A[idx] - stats[c]) * stats[DOUT + c];
}

// ---------------- per-layer driver ----------------
template <int DIN, int DOUT>
static void run_layer(const float* h, const float* W, const int* adj_col,
                      const float* adj_vals, const int* rstart, float* out,
                      char* ws, hipStream_t stream, int N) {
    float* S     = (float*)(ws + SUP_OFF);
    float* A     = (float*)(ws + ACT_OFF);
    float* P     = (float*)(ws + PART_OFF);
    float* stats = (float*)(ws + STATS_OFF);

    int nRT = N / 16;  // N = 100000 -> 6250 tiles exactly
    gemm_wmma<DIN, DOUT><<<(nRT + 7) / 8, 256, 0, stream>>>(h, W, S, nRT);

    spmm_act<DOUT><<<N, DOUT, 0, stream>>>(S, adj_col, adj_vals, rstart, A);

    constexpr int NCH = 250;
    bn_partial<DOUT><<<NCH, 256, 0, stream>>>(A, P, N, NCH);
    bn_finalize<DOUT><<<1, DOUT, 0, stream>>>(P, stats, N, NCH);

    int total = N * DOUT;
    bn_apply<DOUT><<<(total + 255) / 256, 256, 0, stream>>>(A, stats, out, total);
}

extern "C" void kernel_launch(void* const* d_in, const int* in_sizes, int n_in,
                              void* d_out, int out_size, void* d_ws, size_t ws_size,
                              hipStream_t stream) {
    const float* x        = (const float*)d_in[0];
    const int*   adj_row  = (const int*)d_in[1];
    const int*   adj_col  = (const int*)d_in[2];
    const float* adj_vals = (const float*)d_in[3];
    const float* W1       = (const float*)d_in[4];
    const float* W2       = (const float*)d_in[5];
    const float* W3       = (const float*)d_in[6];
    const float* W4       = (const float*)d_in[7];

    const int N = in_sizes[0] / 256;  // 100000
    const int E = in_sizes[1];        // 1600000

    char* ws = (char*)d_ws;
    int* rstart = (int*)(ws + RSTART_OFF);
    row_bounds<<<(N + 1 + 255) / 256, 256, 0, stream>>>(adj_row, rstart, N, E);

    float* out = (float*)d_out;
    float* h1 = out;
    float* h2 = h1 + (size_t)N * 128;
    float* h3 = h2 + (size_t)N * 128;
    float* h4 = h3 + (size_t)N * 64;

    run_layer<256, 128>(x,  W1, adj_col, adj_vals, rstart, h1, ws, stream, N);
    run_layer<128, 128>(h1, W2, adj_col, adj_vals, rstart, h2, ws, stream, N);
    run_layer<128, 64 >(h2, W3, adj_col, adj_vals, rstart, h3, ws, stream, N);
    run_layer<64,  32 >(h3, W4, adj_col, adj_vals, rstart, h4, ws, stream, N);
}
